// AMM_15126874817238
// MI455X (gfx1250) — matrix operand order repaired and checked by reading, not compiled, not run on hardware
//
#include <hip/hip_runtime.h>
#include <hip/hip_bf16.h>
#include <math.h>

// ---------------------------------------------------------------------------
// CDNA5 (gfx1250) wave32 WMMA helpers: V_WMMA_F32_16X16X32_F16
// Fragment k order: k = 16*(lane/16) + s in both A and B (contiguous, not the ISA table order).
// ---------------------------------------------------------------------------
typedef _Float16 h16;
typedef __attribute__((ext_vector_type(16))) _Float16 v16h;
typedef __attribute__((ext_vector_type(8)))  float    v8f;

#define DEVINL __device__ __forceinline__

// Tensor Data Mover availability (ROCm 7.2 lane: 5-arg builtin, no TDM header).
#if defined(__HIP_DEVICE_COMPILE__) && \
    __has_builtin(__builtin_amdgcn_tensor_load_to_lds) && \
    __has_builtin(__builtin_amdgcn_s_wait_tensorcnt) && \
    !__has_include(<hip/amd_detail/amd_gfx1250_TDM.h>)
#define USE_TDM 1
typedef __attribute__((ext_vector_type(4))) unsigned int v4u;
typedef __attribute__((ext_vector_type(8))) int          v8i;
typedef __attribute__((ext_vector_type(4))) int          v4i;
#else
#define USE_TDM 0
#endif

// A-matrix 16x32 f16, row-major source with leading dim `ld` (elements).
DEVINL v16h wmma_load_a(const h16* __restrict__ p, int ld, int lane) {
  int row = lane & 15, half = lane >> 4;
  const h16* rp = p + (size_t)row * ld;
  v16h a;
#pragma unroll
  for (int j = 0; j < 8; ++j) {
    int k = 16 * half + 2 * j;
    a[2 * j]     = rp[k];
    a[2 * j + 1] = rp[k + 1];
  }
  return a;
}

// B-matrix 32x16 f16 from a [col][k] (N-major, K-contiguous) source.
DEVINL v16h wmma_load_bt(const h16* __restrict__ p, int ld, int lane) {
  int col = lane & 15, half = lane >> 4;
  const h16* cp = p + (size_t)col * ld + 16 * half;
  v16h b;
#pragma unroll
  for (int e = 0; e < 16; ++e) b[e] = cp[e];
  return b;
}

DEVINL v8f wmma_f16(v16h a, v16h b, v8f c) {
  return __builtin_amdgcn_wmma_f32_16x16x32_f16(false, a, false, b,
                                                (short)0, c, false, false);
}

// Model constants
#define BB 8
#define CC 128
#define HH 256
#define OCC 256
#define HEADS 4

// ---------------------------------------------------------------------------
__global__ void k_f32_to_f16(const float* __restrict__ s, h16* __restrict__ d, int n) {
  int i = blockIdx.x * blockDim.x + threadIdx.x;
  if (i < n) d[i] = (h16)s[i];
}

// ---------------------------------------------------------------------------
// CPB relative-bias table: bias[h][c][d] = sigmoid(W2 @ relu(W1*rel + b1))
// ---------------------------------------------------------------------------
__global__ void k_bias_tab(const float* __restrict__ w1, const float* __restrict__ b1,
                           const float* __restrict__ w2, float* __restrict__ bias) {
  int i = blockIdx.x * blockDim.x + threadIdx.x;   // 0..16383
  int c = i >> 7, d = i & 127;
  float cd = (float)(d - c) / 127.0f * 8.0f;
  float sg = (cd > 0.f) ? 1.f : ((cd < 0.f) ? -1.f : 0.f);
  float rel = sg * log2f(fabsf(cd) + 1.0f) * (1.0f / 3.0f);   // /log2(8)
  float acc[HEADS] = {0.f, 0.f, 0.f, 0.f};
  for (int j = 0; j < 64; ++j) {
    float h1 = fmaxf(rel * w1[j] + b1[j], 0.f);
#pragma unroll
    for (int h = 0; h < HEADS; ++h) acc[h] += h1 * w2[h * 64 + j];
  }
#pragma unroll
  for (int h = 0; h < HEADS; ++h)
    bias[((size_t)h * 128 + c) * 128 + d] = 1.0f / (1.0f + expf(-acc[h]));
}

// ---------------------------------------------------------------------------
// Fused grouped-1x1 QKV conv (WMMA) + q/k 16x16 maxpool + v depthwise 4x4 s4.
// One block = one (b, 16x16 tile). x tile staged via TDM when available.
// LDS layout (bytes):
//   [0, 131072)        xraw f32 [128 ch][16 y][16 x]  (TDM dest; later cbuf)
//   [0, 65536)         cbuf h16 [256 px][128 c]       (reused after repack)
//   [131072, 204800)   xlds h16 [256 px][XLD]
// ---------------------------------------------------------------------------
#define XLD 144   // padded LDS leading dim (multiple of 16)
#define QKV_SMEM (131072 + 256 * XLD * 2)

__global__ __launch_bounds__(256) void k_qkv(
    const float* __restrict__ x,
    const h16* __restrict__ wq, const float* __restrict__ qb,
    const h16* __restrict__ wk, const float* __restrict__ kb,
    const h16* __restrict__ wv, const float* __restrict__ vb,
    const float* __restrict__ vpw, const float* __restrict__ vpb,
    float* __restrict__ qpool, float* __restrict__ kpool, h16* __restrict__ vT) {
  extern __shared__ char smem[];
  float* xraw = (float*)smem;                   // [128][16][16] f32
  h16* cbuf = (h16*)smem;                       // [256 px][128 c]
  h16* xlds = (h16*)(smem + 131072);            // [256 px][XLD]

  int tid = threadIdx.x, lane = tid & 31, wave = tid >> 5;
  int blk = blockIdx.x;
  int b = blk >> 8, t = blk & 255, ty = t >> 4, tx = t & 15;
  int Y0 = ty * 16, X0 = tx * 16;
  const float* xb = x + (size_t)b * CC * HH * HH;
  const float* gtile = xb + (size_t)Y0 * HH + X0;

#if USE_TDM
  // ---- Tensor Data Mover: DMA 16x16x128 f32 tile into LDS ----
  if (wave == 0) {
    unsigned long long ga = (unsigned long long)(uintptr_t)gtile;
    unsigned int lds_addr = (unsigned int)(uintptr_t)xraw;   // LDS aperture offset
    v4u g0;
    g0[0] = 1u;                                  // count=1, user descriptor
    g0[1] = lds_addr;                            // LDS dest (bytes)
    g0[2] = (unsigned int)(ga & 0xFFFFFFFFull);  // global_addr[31:0]
    g0[3] = (unsigned int)((ga >> 32) & 0x01FFFFFFull) | (2u << 30); // type=2
    v8i g1;
    g1[0] = (int)(2u << 16);                     // data_size = 4 bytes
    g1[1] = (int)(256u << 16);                   // tensor_dim0 = 256 (lo16 @63:48)
    g1[2] = (int)(256u << 16);                   // dim0 hi=0 | tensor_dim1 lo16
    g1[3] = (int)(16u << 16);                    // dim1 hi=0 | tile_dim0 = 16
    g1[4] = (int)(16u | (128u << 16));           // tile_dim1=16, tile_dim2=128
    g1[5] = 256;                                 // tensor_dim0_stride = 256
    g1[6] = 0;                                   // stride0 hi | stride1 lo16 (65536)
    g1[7] = 1;                                   // tensor_dim1_stride >> 16
    v4i g2;
    g2[0] = 128;                                 // tensor_dim2 = 128
    g2[1] = 0; g2[2] = 0; g2[3] = 0;             // tile_dim3 = 0 (3D tile)
    v4i g3 = {0, 0, 0, 0};
    __builtin_amdgcn_tensor_load_to_lds(g0, g1, g2, g3, 0);
    __builtin_amdgcn_s_wait_tensorcnt((short)0);
  }
  __syncthreads();
  // Repack LDS f32 [ch][y][x] -> f16 [px][ch]
  for (int it = 0; it < 32; ++it) {
    int e4 = tid + it * 256;                     // 8192 float4
    int ic = e4 >> 6, rem = e4 & 63;
    int iy = rem >> 2, ix0 = (rem & 3) * 4;
    const float4 v = *(const float4*)(xraw + ic * 256 + iy * 16 + ix0);
    xlds[(iy * 16 + ix0 + 0) * XLD + ic] = (h16)v.x;
    xlds[(iy * 16 + ix0 + 1) * XLD + ic] = (h16)v.y;
    xlds[(iy * 16 + ix0 + 2) * XLD + ic] = (h16)v.z;
    xlds[(iy * 16 + ix0 + 3) * XLD + ic] = (h16)v.w;
  }
#else
  for (int it = 0; it < 32; ++it) {
    int e4 = tid + it * 256;
    int ic = e4 >> 6, rem = e4 & 63;
    int iy = rem >> 2, ix0 = (rem & 3) * 4;
    const float4 v = *(const float4*)(gtile + (size_t)ic * HH * HH + iy * HH + ix0);
    xlds[(iy * 16 + ix0 + 0) * XLD + ic] = (h16)v.x;
    xlds[(iy * 16 + ix0 + 1) * XLD + ic] = (h16)v.y;
    xlds[(iy * 16 + ix0 + 2) * XLD + ic] = (h16)v.z;
    xlds[(iy * 16 + ix0 + 3) * XLD + ic] = (h16)v.w;
  }
#endif
  __syncthreads();

  int g = wave >> 1, mb = wave & 1;             // group 0..3, M-block 0..1
  int half = lane >> 4, col16 = lane & 15;
  for (int m3 = 0; m3 < 3; ++m3) {
    const h16* W = (m3 == 0) ? wq : (m3 == 1) ? wk : wv;
    const float* bi = (m3 == 0) ? qb : (m3 == 1) ? kb : vb;
    v16h a = wmma_load_a(W + (g * 32 + mb * 16) * 32, 32, lane);
    int c0 = g * 32 + mb * 16 + 8 * half;
    for (int nb = 0; nb < 16; ++nb) {
      v8f acc = {0.f, 0.f, 0.f, 0.f, 0.f, 0.f, 0.f, 0.f};
      v16h bf = wmma_load_bt(xlds + (nb * 16) * XLD + g * 32, XLD, lane);
      acc = wmma_f16(a, bf, acc);
      int col = nb * 16 + col16;
      union { h16 h[8]; uint4 u4; } pk;
#pragma unroll
      for (int r = 0; r < 8; ++r) pk.h[r] = (h16)(acc[r] + bi[c0 + r]);
      *(uint4*)(&cbuf[col * 128 + c0]) = pk.u4;   // one b128 LDS store
    }
    __syncthreads();
    if (m3 < 2) {
      float* outp = (m3 == 0) ? qpool : kpool;
      if (tid < CC) {
        float m = -INFINITY;
        for (int p = 0; p < 256; ++p) m = fmaxf(m, (float)cbuf[p * 128 + tid]);
        outp[(((size_t)b * CC + tid) * 16 + ty) * 16 + tx] = m;
      }
    } else {
      for (int it = 0; it < 8; ++it) {
        int e = tid + it * 256;                 // 0..2047
        int c = e >> 4, vp = e & 15;
        int vy = vp >> 2, vx = vp & 3;
        float s = vpb[c];
#pragma unroll
        for (int i2 = 0; i2 < 4; ++i2)
#pragma unroll
          for (int j2 = 0; j2 < 4; ++j2)
            s += (float)cbuf[((vy * 4 + i2) * 16 + (vx * 4 + j2)) * 128 + c] *
                 vpw[c * 16 + i2 * 4 + j2];
        int p = (ty * 4 + vy) * 64 + (tx * 4 + vx);
        vT[((size_t)b * 4096 + p) * CC + c] = (h16)s;      // [b][p][c]
      }
    }
    __syncthreads();
  }
}

// ---------------------------------------------------------------------------
// Depthwise 3x3 on pooled q/k (16x16 -> 14x14) + per-(b,h,c) L2 normalize,
// write f16 padded K=49->64 for WMMA.
// ---------------------------------------------------------------------------
__global__ __launch_bounds__(256) void k_dwnorm(
    const float* __restrict__ qpool, const float* __restrict__ qpw, const float* __restrict__ qpb,
    const float* __restrict__ kpool, const float* __restrict__ kpw, const float* __restrict__ kpb,
    h16* __restrict__ qn, h16* __restrict__ kn) {
  __shared__ float ssum[HEADS];
  int blk = blockIdx.x;
  int mat = blk >> 10;                 // 0=q, 1=k
  int rem = blk & 1023;
  int b = rem >> 7, c = rem & 127;
  const float* pool = ((mat == 0) ? qpool : kpool) + ((size_t)b * CC + c) * 256;
  const float* pw = ((mat == 0) ? qpw : kpw) + c * 9;
  float pb = ((mat == 0) ? qpb : kpb)[c];
  h16* out = (mat == 0) ? qn : kn;

  int tid = threadIdx.x;
  if (tid < HEADS) ssum[tid] = 0.f;
  __syncthreads();
  float val = 0.f;
  int head = 0, k = 0;
  if (tid < 196) {
    int oy = tid / 14, ox = tid % 14;
    float s = pb;
#pragma unroll
    for (int dy = 0; dy < 3; ++dy)
#pragma unroll
      for (int dx = 0; dx < 3; ++dx)
        s += pool[(oy + dy) * 16 + (ox + dx)] * pw[dy * 3 + dx];
    val = s; head = tid / 49; k = tid % 49;
    atomicAdd(&ssum[head], val * val);
  }
  __syncthreads();
  if (tid < 196) {
    float nrm = fmaxf(sqrtf(ssum[head]), 1e-12f);
    out[(((size_t)(b * HEADS + head)) * CC + c) * 64 + k] = (h16)(val / nrm);
  } else {
    int j = tid - 196;
    if (j < 60) {
      int h = j / 15, kk = 49 + j % 15;
      out[(((size_t)(b * HEADS + h)) * CC + c) * 64 + kk] = (h16)0.f;
    }
  }
}

// ---------------------------------------------------------------------------
// sim = scale * qn @ kn^T + bias; double softmax over d; f16 result.
// ---------------------------------------------------------------------------
__global__ __launch_bounds__(256) void k_attn_sim(
    const h16* __restrict__ qn, const h16* __restrict__ kn,
    const float* __restrict__ biasTab, const float* __restrict__ logit_scale,
    h16* __restrict__ simh) {
  extern __shared__ char smem[];
  float* simbuf = (float*)smem;               // [128][128]
  int bh = blockIdx.x;
  int h = bh & 3;
  int tid = threadIdx.x, lane = tid & 31, wave = tid >> 5;
  float scale = expf(fminf(logit_scale[h], logf(100.0f)));
  const h16* q = qn + (size_t)bh * CC * 64;
  const h16* kk = kn + (size_t)bh * CC * 64;
  const float* bt = biasTab + (size_t)h * CC * CC;

  int mb = wave;
  int half = lane >> 4;
  for (int nb = 0; nb < 8; ++nb) {
    v8f acc = {0.f, 0.f, 0.f, 0.f, 0.f, 0.f, 0.f, 0.f};
#pragma unroll
    for (int ks = 0; ks < 2; ++ks) {
      v16h a = wmma_load_a(q + (mb * 16) * 64 + ks * 32, 64, lane);
      v16h bf = wmma_load_bt(kk + (nb * 16) * 64 + ks * 32, 64, lane);
      acc = wmma_f16(a, bf, acc);
    }
    int d = nb * 16 + (lane & 15);
#pragma unroll
    for (int r = 0; r < 8; ++r) {
      int c = mb * 16 + r + 8 * half;
      simbuf[c * 128 + d] = acc[r] * scale + bt[c * 128 + d];
    }
  }
  __syncthreads();
  if (tid < 128) {
    float* row = simbuf + tid * 128;
    float m1 = -INFINITY;
    for (int i = 0; i < 128; ++i) m1 = fmaxf(m1, row[i]);
    float s1 = 0.f;
    for (int i = 0; i < 128; ++i) s1 += expf(row[i] - m1);
    float inv1 = 1.0f / s1;
    float pmin = INFINITY;                     // max(1-p) = 1-min(p)
    for (int i = 0; i < 128; ++i) pmin = fminf(pmin, expf(row[i] - m1) * inv1);
    float s2 = 0.f;
    for (int i = 0; i < 128; ++i) s2 += expf(pmin - expf(row[i] - m1) * inv1);
    float inv2 = 1.0f / s2;
    h16* orow = simh + (size_t)bh * CC * CC + tid * 128;
    for (int i = 0; i < 128; ++i)
      orow[i] = (h16)(expf(pmin - expf(row[i] - m1) * inv1) * inv2);
  }
}

// ---------------------------------------------------------------------------
// out = sim(128x128) @ v(128x1024) per (b,h); sim staged in LDS once/block.
// Grid = (b*h) x 8 N-chunks of 128; writes Y1 [b][p][c] with packed b128.
// ---------------------------------------------------------------------------
__global__ __launch_bounds__(256) void k_attn_av(
    const h16* __restrict__ simh, const h16* __restrict__ vT, h16* __restrict__ Y1) {
  extern __shared__ char smem[];
  h16* sS = (h16*)smem;                        // [128][128] f16 (32 KB)
  int blk = blockIdx.x;
  int bh = blk >> 3, nch = blk & 7;
  int b = bh >> 2, h = bh & 3;
  int tid = threadIdx.x, lane = tid & 31, wave = tid >> 5;
  const h16* S = simh + (size_t)bh * CC * CC;
  const h16* V = vT + ((size_t)b * 4096 + h * 1024 + nch * 128) * CC;
  __builtin_prefetch(V, 0, 0);
  // Stage sim into LDS (2048 uint4)
  for (int it = 0; it < 8; ++it) {
    int e = tid + it * 256;
    ((uint4*)sS)[e] = ((const uint4*)S)[e];
  }
  __syncthreads();
  int n0 = wave * 16;
  int half = lane >> 4;
#pragma unroll 1
  for (int mb = 0; mb < 8; ++mb) {
    v8f acc = {0.f, 0.f, 0.f, 0.f, 0.f, 0.f, 0.f, 0.f};
#pragma unroll 1
    for (int ks = 0; ks < 8; ++ks) {
      v16h a = wmma_load_a(sS + (mb * 16) * 128 + ks * 32, 128, lane);
      v16h bf = wmma_load_bt(V + (size_t)n0 * 128 + ks * 32, 128, lane);
      acc = wmma_f16(a, bf, acc);
    }
    int p = h * 1024 + nch * 128 + n0 + (lane & 15);
    int c0 = mb * 16 + 8 * half;
    union { h16 hh[8]; uint4 u4; } pk;
#pragma unroll
    for (int r = 0; r < 8; ++r) pk.hh[r] = (h16)acc[r];
    *(uint4*)(&Y1[((size_t)b * 4096 + p) * CC + c0]) = pk.u4;
  }
}

// ---------------------------------------------------------------------------
// proj1 (+GELU) -> proj2 -> + shortcut GEMM (K=2048) -> dual channel-LN.
// ---------------------------------------------------------------------------
__global__ __launch_bounds__(256) void k_proj(
    const h16* __restrict__ Y1, const h16* __restrict__ W1h, const float* __restrict__ b1,
    const h16* __restrict__ W2h, const float* __restrict__ b2,
    const h16* __restrict__ WSCh, const float* __restrict__ scb,
    const float* __restrict__ ng, const float* __restrict__ nbe,
    const float* __restrict__ sg, const float* __restrict__ sbe,
    const float* __restrict__ x, float* __restrict__ out) {
  extern __shared__ char smem[];
  h16* xpatch = (h16*)smem;                              // [16 px][2048 k]
  h16* g1buf = (h16*)(smem + 65536);                     // [16 px][128]
  float* z2buf = (float*)(smem + 65536 + 4096);          // [16 px][256]
  float* sbuf  = (float*)(smem + 65536 + 4096 + 16384);  // [16 px][256]

  int blk = blockIdx.x;
  int b = blk >> 8, tile = blk & 255;
  int px0 = tile * 16;
  int oy = px0 >> 6, ox0 = px0 & 63;                     // 16 px share one row
  int tid = threadIdx.x, lane = tid & 31, wave = tid >> 5;
  int half = lane >> 4, col = lane & 15;

  // Stage shortcut patches (float4 loads, 8-byte packed LDS stores):
  // x[b][ic][4oy+iy][4ox0+ixx] -> xpatch[px][ic*16 + iy*4 + jx]
  const float* xb = x + (size_t)b * CC * HH * HH + (size_t)(4 * oy) * HH + 4 * ox0;
  for (int it = 0; it < 32; ++it) {
    int e4 = tid + it * 256;                   // 8192 float4
    int ic = e4 >> 6, rem = e4 & 63;
    int iy = rem >> 4, ix0 = (rem & 15) * 4;
    const float4 v = *(const float4*)(xb + (size_t)ic * HH * HH + iy * HH + ix0);
    union { h16 hh[4]; uint2 u2; } pk;
    pk.hh[0] = (h16)v.x; pk.hh[1] = (h16)v.y; pk.hh[2] = (h16)v.z; pk.hh[3] = (h16)v.w;
    *(uint2*)(&xpatch[(ix0 >> 2) * 2048 + ic * 16 + iy * 4]) = pk.u2;
  }
  __syncthreads();

  // GEMM1: 128x16 = W1(128x128) @ Y1tile, then exact GELU -> LDS f16
  {
    int mb = wave;
    v8f acc = {0.f, 0.f, 0.f, 0.f, 0.f, 0.f, 0.f, 0.f};
    const h16* Bb = Y1 + ((size_t)b * 4096 + px0) * CC;
#pragma unroll
    for (int ks = 0; ks < 4; ++ks) {
      v16h a = wmma_load_a(W1h + (mb * 16) * 128 + ks * 32, 128, lane);
      v16h bf = wmma_load_bt(Bb + ks * 32, 128, lane);
      acc = wmma_f16(a, bf, acc);
    }
    int oc0 = mb * 16 + 8 * half;
    union { h16 hh[8]; uint4 u4; } pk;
#pragma unroll
    for (int r = 0; r < 8; ++r) {
      float z = acc[r] + b1[oc0 + r];
      pk.hh[r] = (h16)(0.5f * z * (1.0f + erff(z * 0.70710678118f)));
    }
    *(uint4*)(&g1buf[col * 128 + oc0]) = pk.u4;
  }
  __syncthreads();

  // GEMM2: 256x16 = W2(256x128) @ gelu -> z2buf (f32)
  for (int mi = 0; mi < 2; ++mi) {
    int mb = wave + mi * 8;
    v8f acc = {0.f, 0.f, 0.f, 0.f, 0.f, 0.f, 0.f, 0.f};
#pragma unroll
    for (int ks = 0; ks < 4; ++ks) {
      v16h a = wmma_load_a(W2h + (mb * 16) * 128 + ks * 32, 128, lane);
      v16h bf = wmma_load_bt(g1buf + ks * 32, 128, lane);
      acc = wmma_f16(a, bf, acc);
    }
    int oc0 = mb * 16 + 8 * half;
    float4 lo = make_float4(acc[0] + b2[oc0 + 0], acc[1] + b2[oc0 + 1],
                            acc[2] + b2[oc0 + 2], acc[3] + b2[oc0 + 3]);
    float4 hi = make_float4(acc[4] + b2[oc0 + 4], acc[5] + b2[oc0 + 5],
                            acc[6] + b2[oc0 + 6], acc[7] + b2[oc0 + 7]);
    *(float4*)(&z2buf[col * 256 + oc0]) = lo;
    *(float4*)(&z2buf[col * 256 + oc0 + 4]) = hi;
  }

  // Shortcut GEMM: 256x16 = Wsc(256x2048) @ xpatch -> sbuf (f32)
  for (int mi = 0; mi < 2; ++mi) {
    int mb = wave + mi * 8;
    const h16* Wrow = WSCh + (size_t)(mb * 16) * 2048;
    __builtin_prefetch(Wrow, 0, 0);
    v8f acc = {0.f, 0.f, 0.f, 0.f, 0.f, 0.f, 0.f, 0.f};
#pragma unroll 1
    for (int ks = 0; ks < 64; ++ks) {
      v16h a = wmma_load_a(Wrow + ks * 32, 2048, lane);
      v16h bf = wmma_load_bt(xpatch + ks * 32, 2048, lane);
      acc = wmma_f16(a, bf, acc);
    }
    int oc0 = mb * 16 + 8 * half;
    float4 lo = make_float4(acc[0] + scb[oc0 + 0], acc[1] + scb[oc0 + 1],
                            acc[2] + scb[oc0 + 2], acc[3] + scb[oc0 + 3]);
    float4 hi = make_float4(acc[4] + scb[oc0 + 4], acc[5] + scb[oc0 + 5],
                            acc[6] + scb[oc0 + 6], acc[7] + scb[oc0 + 7]);
    *(float4*)(&sbuf[col * 256 + oc0]) = lo;
    *(float4*)(&sbuf[col * 256 + oc0 + 4]) = hi;
  }
  __syncthreads();

  // Dual channel-LayerNorm + sum, one owner thread per pixel
  if (tid < 16) {
    int pl = tid;
    float s = 0.f, s2 = 0.f, t = 0.f, t2 = 0.f;
    for (int oc = 0; oc < 256; ++oc) {
      float a = z2buf[pl * 256 + oc]; s += a;  s2 += a * a;
      float c2 = sbuf[pl * 256 + oc]; t += c2; t2 += c2 * c2;
    }
    float mu = s * (1.0f / 256.0f),  var  = s2 * (1.0f / 256.0f) - mu * mu;
    float mu2 = t * (1.0f / 256.0f), var2 = t2 * (1.0f / 256.0f) - mu2 * mu2;
    float r1 = rsqrtf(var + 1e-5f), r2 = rsqrtf(var2 + 1e-5f);
    int p = px0 + pl;
    for (int oc = 0; oc < 256; ++oc) {
      float a = (z2buf[pl * 256 + oc] - mu) * r1 * ng[oc] + nbe[oc];
      float c2 = (sbuf[pl * 256 + oc] - mu2) * r2 * sg[oc] + sbe[oc];
      out[((size_t)b * OCC + oc) * 4096 + p] = a + c2;
    }
  }
}

// ---------------------------------------------------------------------------
extern "C" void kernel_launch(void* const* d_in, const int* in_sizes, int n_in,
                              void* d_out, int out_size, void* d_ws, size_t ws_size,
                              hipStream_t stream) {
  (void)in_sizes; (void)n_in; (void)out_size; (void)ws_size;
  const float* x       = (const float*)d_in[0];
  const float* q_w     = (const float*)d_in[1];
  const float* q_b     = (const float*)d_in[2];
  const float* k_w     = (const float*)d_in[3];
  const float* k_b     = (const float*)d_in[4];
  const float* v_w     = (const float*)d_in[5];
  const float* v_b     = (const float*)d_in[6];
  const float* qp_w    = (const float*)d_in[7];
  const float* qp_b    = (const float*)d_in[8];
  const float* kp_w    = (const float*)d_in[9];
  const float* kp_b    = (const float*)d_in[10];
  const float* vp_w    = (const float*)d_in[11];
  const float* vp_b    = (const float*)d_in[12];
  const float* lscale  = (const float*)d_in[13];
  const float* cpb_w1  = (const float*)d_in[14];
  const float* cpb_b1  = (const float*)d_in[15];
  const float* cpb_w2  = (const float*)d_in[16];
  const float* proj_w1 = (const float*)d_in[17];
  const float* proj_b1 = (const float*)d_in[18];
  const float* proj_w2 = (const float*)d_in[19];
  const float* proj_b2 = (const float*)d_in[20];
  const float* norm_g  = (const float*)d_in[21];
  const float* norm_be = (const float*)d_in[22];
  const float* sc_w    = (const float*)d_in[23];
  const float* sc_cb   = (const float*)d_in[24];
  const float* sc_g    = (const float*)d_in[25];
  const float* sc_be   = (const float*)d_in[26];

  char* ws = (char*)d_ws;
  size_t off = 0;
  auto alloc = [&](size_t bytes) -> char* {
    char* p = ws + off;
    off += (bytes + 255) & ~(size_t)255;
    return p;
  };
  float* BIAS  = (float*)alloc((size_t)HEADS * 128 * 128 * 4);
  h16*   WQH   = (h16*)alloc(4096 * 2);
  h16*   WKH   = (h16*)alloc(4096 * 2);
  h16*   WVH   = (h16*)alloc(4096 * 2);
  h16*   W1H   = (h16*)alloc(16384 * 2);
  h16*   W2H   = (h16*)alloc(32768 * 2);
  h16*   WSCH  = (h16*)alloc(524288 * 2);
  float* QPOOL = (float*)alloc((size_t)BB * CC * 256 * 4);
  float* KPOOL = (float*)alloc((size_t)BB * CC * 256 * 4);
  h16*   QN    = (h16*)alloc((size_t)BB * HEADS * CC * 64 * 2);
  h16*   KN    = (h16*)alloc((size_t)BB * HEADS * CC * 64 * 2);
  h16*   VT    = (h16*)alloc((size_t)BB * 4096 * CC * 2);
  h16*   SIMH  = (h16*)alloc((size_t)BB * HEADS * CC * CC * 2);
  h16*   Y1    = (h16*)alloc((size_t)BB * 4096 * CC * 2);

  // Weight precision conversion (f32 -> f16)
  k_f32_to_f16<<<16, 256, 0, stream>>>(q_w, WQH, 4096);
  k_f32_to_f16<<<16, 256, 0, stream>>>(k_w, WKH, 4096);
  k_f32_to_f16<<<16, 256, 0, stream>>>(v_w, WVH, 4096);
  k_f32_to_f16<<<64, 256, 0, stream>>>(proj_w1, W1H, 16384);
  k_f32_to_f16<<<128, 256, 0, stream>>>(proj_w2, W2H, 32768);
  k_f32_to_f16<<<2048, 256, 0, stream>>>(sc_w, WSCH, 524288);

  // Input-independent CPB bias table
  k_bias_tab<<<64, 256, 0, stream>>>(cpb_w1, cpb_b1, cpb_w2, BIAS);

  // Fused QKV + pooling (TDM-staged tiles + WMMA grouped conv)
  k_qkv<<<2048, 256, QKV_SMEM, stream>>>(
      x, WQH, q_b, WKH, k_b, WVH, v_b, vp_w, vp_b, QPOOL, KPOOL, VT);

  // Depthwise 3x3 + head-wise L2 norm
  k_dwnorm<<<2048, 256, 0, stream>>>(QPOOL, qp_w, qp_b, KPOOL, kp_w, kp_b, QN, KN);

  // sim + double softmax
  k_attn_sim<<<BB * HEADS, 256, 128 * 128 * 4, stream>>>(QN, KN, BIAS, lscale, SIMH);

  // attention output GEMM (sim staged in LDS)
  k_attn_av<<<BB * HEADS * 8, 256, 32768, stream>>>(SIMH, VT, Y1);

  // projections + shortcut + LayerNorms
  k_proj<<<2048, 256, 65536 + 4096 + 16384 + 16384, stream>>>(
      Y1, W1H, proj_b1, W2H, proj_b2, WSCH, sc_cb,
      norm_g, norm_be, sc_g, sc_be, x, (float*)d_out);

  // Second tuple output: verbatim copy of x
  hipMemcpyAsync((float*)d_out + (size_t)BB * OCC * 4096, x,
                 (size_t)BB * CC * HH * HH * sizeof(float),
                 hipMemcpyDeviceToDevice, stream);
}
